// ConvolutionalSelfAttentionFusion_54735063220775
// MI455X (gfx1250) — compile-verified, matrix operand order checked by reading, not run on hardware
//
#include <hip/hip_runtime.h>
#include <hip/hip_bf16.h>
#include <stdint.h>

// ---------------------------------------------------------------------------
// ConvolutionalSelfAttentionFusion for MI455X (gfx1250, wave32, WMMA + TDM)
// B=8, C=256, H=W=96, K=3
// Pipeline: f32->bf16 convert (+weight transpose) |
//           3x conv (bf16 WMMA implicit GEMM, double-buffered LDS, TDM weights)
//           per-(b,c) row attention (WMMA + wave32 shfl softmax) | final conv
// ---------------------------------------------------------------------------

#define BATCH 8
#define CH    256
#define HH    96
#define WW    96
#define HW    (HH * WW)          // 9216
#define NE    (BATCH * CH * HW)  // 18,874,368 elements per tensor
#define NWT   (CH * CH * 9)      // 589,824 weight elements per conv

typedef __attribute__((ext_vector_type(16))) __bf16 v16bf;
typedef __attribute__((ext_vector_type(8)))  __bf16 v8bf;
typedef __attribute__((ext_vector_type(8)))  float  v8f;

#if (__has_builtin(__builtin_amdgcn_tensor_load_to_lds) && \
     __has_builtin(__builtin_amdgcn_s_wait_tensorcnt))
#define HAS_TDM 1
#else
#define HAS_TDM 0
#endif

// Build a 16x16x32 bf16 A/B fragment from LDS: per lane, 8 contiguous bf16 at
// k = base..base+7 and 8 more at k = base+16..base+23 (ISA 7.12.2 layout).
__device__ __forceinline__ v16bf load_frag(const __bf16* p) {
  v8bf lo = *(const v8bf*)(p);
  v8bf hi = *(const v8bf*)(p + 16);
  v16bf r;
#pragma unroll
  for (int j = 0; j < 8; ++j) { r[j] = lo[j]; r[8 + j] = hi[j]; }
  return r;
}

__device__ __forceinline__ v8f wmma_bf16(v16bf a, v16bf b, v8f c) {
  return __builtin_amdgcn_wmma_f32_16x16x32_bf16(
      false, a, false, b, (short)0, c, false, false);
}

#if HAS_TDM
typedef unsigned int v4u __attribute__((ext_vector_type(4)));
typedef int          v4i __attribute__((ext_vector_type(4)));
typedef int          v8i __attribute__((ext_vector_type(8)));

__device__ __forceinline__ uint32_t lds_offset_of(const void* p) {
  return (uint32_t)(uintptr_t)(__attribute__((address_space(3))) const void*)p;
}

// One 2D tile: 576 rows x 32 bf16 (64B), row stride 256 bf16 (512B) -> packed LDS.
__device__ __forceinline__ void tdm_load_weights(const __bf16* gsrc,
                                                 uint32_t lds_byte_off) {
  const uint64_t ga = (uint64_t)(uintptr_t)gsrc;
  v4u g0;
  g0[0] = 1u;                                              // count=1, user mode
  g0[1] = lds_byte_off;                                    // lds_addr
  g0[2] = (uint32_t)ga;                                    // global_addr[31:0]
  g0[3] = (uint32_t)((ga >> 32) & 0x1ffffffu) | (2u << 30); // addr[56:32] | type=2
  v8i g1;
  g1[0] = (int)(1u << 16);        // data_size = 2 bytes; mask=0; no pad/iterate
  g1[1] = (int)(256u << 16);      // tensor_dim0 = 256 (low 16 in [63:48])
  g1[2] = (int)(576u << 16);      // tensor_dim0 hi = 0 | tensor_dim1 = 576
  g1[3] = (int)(32u << 16);       // tensor_dim1 hi = 0 | tile_dim0 = 32
  g1[4] = 576;                    // tile_dim1 = 576, tile_dim2 = 0
  g1[5] = 256;                    // tensor_dim0_stride = 256 (low 32)
  g1[6] = 0;
  g1[7] = 0;
  const v4i z4 = {0, 0, 0, 0};
  const v8i z8 = {0, 0, 0, 0, 0, 0, 0, 0};
  // 6-arg toolchain variant: (g0, g1, g2, g3, extra, cpol)
  __builtin_amdgcn_tensor_load_to_lds(g0, g1, z4, z4, z8, 0);
}
#endif

// ---------------------------------------------------------------------------
// f32 -> bf16 conversions
// ---------------------------------------------------------------------------
__global__ void cvt_f32_bf16(const float* __restrict__ src,
                             __bf16* __restrict__ dst, int n) {
  int i = blockIdx.x * blockDim.x + threadIdx.x;
  if (i < n) dst[i] = (__bf16)__builtin_nontemporal_load(&src[i]);
}

// weights [co][ci][tap] (f32) -> [co][tap][ci] (bf16): TDM/fragment friendly
__global__ void cvt_wt_transpose(const float* __restrict__ w,
                                 __bf16* __restrict__ o) {
  int idx = blockIdx.x * blockDim.x + threadIdx.x;
  if (idx >= NWT) return;
  const int co = idx / 2304;
  const int r  = idx - co * 2304;
  const int tap = r >> 8;
  const int ci  = r & 255;
  o[idx] = (__bf16)__builtin_nontemporal_load(&w[((size_t)co * CH + ci) * 9 + tap]);
}

// ---------------------------------------------------------------------------
// 3x3 SAME conv as implicit GEMM, double-buffered LDS, TDM weight staging.
// grid = (4 co-groups, 96 rows, 8 batches), block = 256 threads (8 waves).
// ---------------------------------------------------------------------------
template <bool OUT_F32>
__launch_bounds__(256, 2)
__global__ void conv3x3_wmma(const __bf16* __restrict__ in,   // [C][96][96] per b
                             const __bf16* __restrict__ wtT,  // [co][tap][ci]
                             const float*  __restrict__ bias, // [C]
                             void* __restrict__ out) {
  __shared__ __attribute__((aligned(16))) __bf16 sIn[2][3][104][32]; // [kh][padcol][ci]
  __shared__ __attribute__((aligned(16))) __bf16 sW[2][64][9][32];   // [co][tap][ci]

  const int tid  = threadIdx.x;
  const int co0  = blockIdx.x * 64;
  const int h    = blockIdx.y;
  const int b    = blockIdx.z;
  const int lane = tid & 31;
  const int wave = tid >> 5;
  const int wm   = wave & 3;   // co sub-tile (16 rows)
  const int wn   = wave >> 2;  // col half (0 -> 0..47, 1 -> 48..95)
  const int ln   = lane & 15;
  const int half = lane >> 4;

  const __bf16* inB   = in  + (size_t)b * CH * HW;
  const __bf16* wBase = wtT + (size_t)co0 * 9 * CH;  // 576 rows of 256 ci

  // --- staging helpers -----------------------------------------------------
  v8bf val[5];  // input prefetch registers: 1152 16B-chunks / 256 threads

  auto load_input_regs = [&](int chunk) {
    const int ci0 = chunk * 32;
#pragma unroll
    for (int s = 0; s < 5; ++s) {
#pragma unroll
      for (int e = 0; e < 8; ++e) val[s][e] = (__bf16)0.0f;
      const int i = tid + s * 256;
      if (i < 1152) {
        const int kh = i / 384, rem = i - kh * 384;
        const int ci = rem / 12, g = rem - ci * 12;
        const int r = h + kh - 1;
        if (r >= 0 && r < HH)
          val[s] = *(const v8bf*)(inB + (size_t)(ci0 + ci) * HW + r * WW + g * 8);
      }
    }
  };

  auto store_input_regs = [&](int buf) {
#pragma unroll
    for (int s = 0; s < 5; ++s) {
      const int i = tid + s * 256;
      if (i < 1152) {
        const int kh = i / 384, rem = i - kh * 384;
        const int ci = rem / 12, g = rem - ci * 12;
#pragma unroll
        for (int e = 0; e < 8; ++e) sIn[buf][kh][g * 8 + e + 1][ci] = val[s][e];
      }
    }
    if (tid < 96) {  // halo columns 0 and 97
      const int kh = tid >> 5, j = tid & 31;
      uint32_t* p = (uint32_t*)(j < 16 ? &sIn[buf][kh][0][0] : &sIn[buf][kh][97][0]);
      p[j & 15] = 0u;
    }
  };

#if !HAS_TDM
  auto stage_weights_manual = [&](int buf, int chunk) {
    const int ci0 = chunk * 32;
    __bf16* dst = &sW[buf][0][0][0];
    for (int c = tid; c < 2304; c += 256) {       // 576 rows x 4 16B quarters
      const int row = c >> 2, q = c & 3;
      *(v8bf*)(dst + row * 32 + q * 8) =
          *(const v8bf*)(wBase + (size_t)row * CH + ci0 + q * 8);
    }
  };
#endif

  v8f acc[3];
#pragma unroll
  for (int t = 0; t < 3; ++t)
#pragma unroll
    for (int j = 0; j < 8; ++j) acc[t][j] = 0.0f;

  // --- prologue: stage chunk 0 into buffer 0 ------------------------------
  load_input_regs(0);
#if HAS_TDM
  if (wave == 0) tdm_load_weights(wBase, lds_offset_of(&sW[0][0][0][0]));
#else
  stage_weights_manual(0, 0);
#endif
  store_input_regs(0);
#if HAS_TDM
  if (wave == 0) __builtin_amdgcn_s_wait_tensorcnt(0);
#endif
  __syncthreads();

  // --- main K loop: 8 chunks of 32 ci, ping-pong buffers -------------------
  for (int c = 0; c < 8; ++c) {
    const int cur = c & 1, nxt = cur ^ 1;
    const bool more = (c < 7);
    if (more) {
      load_input_regs(c + 1);  // global loads in flight during compute
#if HAS_TDM
      if (wave == 0)
        tdm_load_weights(wBase + (c + 1) * 32, lds_offset_of(&sW[nxt][0][0][0]));
#endif
    }

#pragma unroll
    for (int kh = 0; kh < 3; ++kh) {
#pragma unroll
      for (int kw = 0; kw < 3; ++kw) {
        const v16bf a = load_frag(&sW[cur][wm * 16 + ln][kh * 3 + kw][half * 8]);
#pragma unroll
        for (int t16 = 0; t16 < 3; ++t16) {
          const int pcol = wn * 48 + t16 * 16 + ln + kw;
          const v16bf bf = load_frag(&sIn[cur][kh][pcol][half * 8]);
          acc[t16] = wmma_bf16(a, bf, acc[t16]);
        }
      }
    }

    if (more) {
      store_input_regs(nxt);
#if !HAS_TDM
      stage_weights_manual(nxt, c + 1);
#endif
    }
#if HAS_TDM
    if (wave == 0) __builtin_amdgcn_s_wait_tensorcnt(0);
#endif
    __syncthreads();
  }

  // --- epilogue: bias add, store -------------------------------------------
#pragma unroll
  for (int t16 = 0; t16 < 3; ++t16) {
#pragma unroll
    for (int i = 0; i < 8; ++i) {
      const int co  = co0 + wm * 16 + half * 8 + i;
      const int col = wn * 48 + t16 * 16 + ln;
      const float v = acc[t16][i] + bias[co];
      const size_t oidx = ((size_t)b * CH + co) * HW + h * WW + col;
      if (OUT_F32) __builtin_nontemporal_store(v, (float*)out + oidx);
      else         ((__bf16*)out)[oidx] = (__bf16)v;
    }
  }
}

// ---------------------------------------------------------------------------
// Row attention per (b,c): S = (Q Kt)/sqrt(W), P = softmax(S), F = P V.
// grid = (C, B), block = 192 threads (6 waves); wave ti owns rows ti*16..+15.
// ---------------------------------------------------------------------------
__launch_bounds__(192)
__global__ void attn_rows_wmma(const __bf16* __restrict__ q,
                               const __bf16* __restrict__ k,
                               const __bf16* __restrict__ v,
                               __bf16* __restrict__ fused) {
  __shared__ __attribute__((aligned(16))) __bf16 sQ[96 * 104];  // reused for P
  __shared__ __attribute__((aligned(16))) __bf16 sK[96 * 104];  // reused for F
  __shared__ __attribute__((aligned(16))) __bf16 sVt[96 * 104]; // [w][j]

  const int tid = threadIdx.x;
  const size_t base = ((size_t)blockIdx.y * CH + blockIdx.x) * HW;

  for (int c = tid; c < 1152; c += 192) {      // 16B chunks of the 96x96 tiles
    const int row = c / 12, g = c - row * 12;
    const v8bf qv = *(const v8bf*)(q + base + row * WW + g * 8);
    const v8bf kv = *(const v8bf*)(k + base + row * WW + g * 8);
    const v8bf vv = *(const v8bf*)(v + base + row * WW + g * 8);
    *(v8bf*)&sQ[row * 104 + g * 8] = qv;
    *(v8bf*)&sK[row * 104 + g * 8] = kv;
#pragma unroll
    for (int e = 0; e < 8; ++e) sVt[(g * 8 + e) * 104 + row] = vv[e];
  }
  __syncthreads();

  const int lane = tid & 31;
  const int ti   = tid >> 5;
  const int ln   = lane & 15;
  const int half = lane >> 4;

  // ---- S = Q * K^T ----
  v8f acc[6];
#pragma unroll
  for (int t = 0; t < 6; ++t)
#pragma unroll
    for (int j = 0; j < 8; ++j) acc[t][j] = 0.0f;

#pragma unroll
  for (int kc = 0; kc < 3; ++kc) {
    const v16bf a = load_frag(&sQ[(ti * 16 + ln) * 104 + kc * 32 + half * 8]);
#pragma unroll
    for (int tj = 0; tj < 6; ++tj) {
      const v16bf bf = load_frag(&sK[(tj * 16 + ln) * 104 + kc * 32 + half * 8]);
      acc[tj] = wmma_bf16(a, bf, acc[tj]);
    }
  }

  // ---- softmax over each row ----
  const float scale = 0.10206207261596577f;  // 1/sqrt(96)
#pragma unroll
  for (int i = 0; i < 8; ++i) {
    float e[6];
    float m = -3.4e38f;
#pragma unroll
    for (int tj = 0; tj < 6; ++tj) { e[tj] = acc[tj][i] * scale; m = fmaxf(m, e[tj]); }
#pragma unroll
    for (int o = 8; o; o >>= 1) m = fmaxf(m, __shfl_xor(m, o));
    float s = 0.0f;
#pragma unroll
    for (int tj = 0; tj < 6; ++tj) { e[tj] = __expf(e[tj] - m); s += e[tj]; }
#pragma unroll
    for (int o = 8; o; o >>= 1) s += __shfl_xor(s, o);
    const float inv = 1.0f / s;
    const int row = ti * 16 + half * 8 + i;
#pragma unroll
    for (int tj = 0; tj < 6; ++tj)
      sQ[row * 104 + tj * 16 + ln] = (__bf16)(e[tj] * inv);  // wave-private rows
  }
  __syncthreads();

  // ---- F = P * V ----
  v8f fa[6];
#pragma unroll
  for (int t = 0; t < 6; ++t)
#pragma unroll
    for (int j = 0; j < 8; ++j) fa[t][j] = 0.0f;

#pragma unroll
  for (int kc = 0; kc < 3; ++kc) {
    const v16bf a = load_frag(&sQ[(ti * 16 + ln) * 104 + kc * 32 + half * 8]);
#pragma unroll
    for (int tw = 0; tw < 6; ++tw) {
      const v16bf bf = load_frag(&sVt[(tw * 16 + ln) * 104 + kc * 32 + half * 8]);
      fa[tw] = wmma_bf16(a, bf, fa[tw]);
    }
  }

  // ---- F -> LDS -> coalesced b128 global stores ----
#pragma unroll
  for (int i = 0; i < 8; ++i) {
    const int row = ti * 16 + half * 8 + i;
#pragma unroll
    for (int tw = 0; tw < 6; ++tw)
      sK[row * 104 + tw * 16 + ln] = (__bf16)fa[tw][i];
  }
  __syncthreads();
  for (int c = tid; c < 1152; c += 192) {
    const int row = c / 12, g = c - row * 12;
    *(v8bf*)(fused + base + row * WW + g * 8) = *(const v8bf*)&sK[row * 104 + g * 8];
  }
}

// ---------------------------------------------------------------------------
// Host launch
// ---------------------------------------------------------------------------
extern "C" void kernel_launch(void* const* d_in, const int* in_sizes, int n_in,
                              void* d_out, int out_size, void* d_ws, size_t ws_size,
                              hipStream_t stream) {
  const float* feats = (const float*)d_in[0];
  const float* graph = (const float*)d_in[1];
  const float* wq = (const float*)d_in[2];
  const float* bq = (const float*)d_in[3];
  const float* wk = (const float*)d_in[4];
  const float* bk = (const float*)d_in[5];
  const float* wv = (const float*)d_in[6];
  const float* bv = (const float*)d_in[7];
  const float* wo = (const float*)d_in[8];
  const float* bo = (const float*)d_in[9];

  __bf16* ws   = (__bf16*)d_ws;
  __bf16* gBf  = ws;
  __bf16* fBf  = gBf + NE;
  __bf16* qBf  = fBf + NE;
  __bf16* kBf  = qBf + NE;
  __bf16* vBf  = kBf + NE;
  __bf16* fuBf = vBf + NE;
  __bf16* wqT  = fuBf + NE;
  __bf16* wkT  = wqT + NWT;
  __bf16* wvT  = wkT + NWT;
  __bf16* woT  = wvT + NWT;

  const int thr = 256;
  cvt_f32_bf16<<<(NE + thr - 1) / thr, thr, 0, stream>>>(graph, gBf, NE);
  cvt_f32_bf16<<<(NE + thr - 1) / thr, thr, 0, stream>>>(feats, fBf, NE);
  cvt_wt_transpose<<<(NWT + thr - 1) / thr, thr, 0, stream>>>(wq, wqT);
  cvt_wt_transpose<<<(NWT + thr - 1) / thr, thr, 0, stream>>>(wk, wkT);
  cvt_wt_transpose<<<(NWT + thr - 1) / thr, thr, 0, stream>>>(wv, wvT);
  cvt_wt_transpose<<<(NWT + thr - 1) / thr, thr, 0, stream>>>(wo, woT);

  const dim3 cgrid(4, HH, BATCH);
  conv3x3_wmma<false><<<cgrid, 256, 0, stream>>>(gBf, wqT, bq, (void*)qBf);
  conv3x3_wmma<false><<<cgrid, 256, 0, stream>>>(fBf, wkT, bk, (void*)kBf);
  conv3x3_wmma<false><<<cgrid, 256, 0, stream>>>(fBf, wvT, bv, (void*)vBf);

  attn_rows_wmma<<<dim3(CH, BATCH), 192, 0, stream>>>(qBf, kBf, vBf, fuBf);

  conv3x3_wmma<true><<<cgrid, 256, 0, stream>>>(fuBf, woT, bo, d_out);
}
